// ScaleDotProduct_80668075753828
// MI455X (gfx1250) — matrix operand order repaired and checked by reading, not compiled, not run on hardware
//
#include <hip/hip_runtime.h>
#include <hip/hip_bf16.h>
#include <math.h>

// ---------------------------------------------------------------------------
// ScaleDotProduct for MI455X (gfx1250, wave32, WMMA bf16 + TDM)
//   B=8, S=2048, INS=768, OUTS=64
//   Phase 0: convert W -> bf16 transposed (tiny); X converted in-kernel
//   Phase 1: QKV projection via v_wmma_f32_16x16x32_bf16 (bias fused).
//            W k-chunks staged into LDS by the TDM (shared by 8 waves,
//            double-buffered); X fp32 streamed with 1-iteration pipeline.
//   Phase 2: flash-style attention; K/V tiles staged into LDS by the TDM,
//            double-buffered, one workgroup barrier per 32-key tile
// ---------------------------------------------------------------------------

#define INS   768
#define OUTS  64
#define BATCH 8
#define SEQ   2048

typedef __attribute__((ext_vector_type(16))) __bf16 v16bf;
typedef __attribute__((ext_vector_type(8)))  __bf16 v8bf;
typedef __attribute__((ext_vector_type(8)))  float  v8f;
typedef __attribute__((ext_vector_type(4)))  unsigned int u32x4;
typedef __attribute__((ext_vector_type(8)))  int    i32x8;
typedef __attribute__((ext_vector_type(4)))  int    i32x4;

// Native bf16 convert: CDNA5 has V_CVT_PK_BF16_F32; clang legalizes this cast.
__device__ __forceinline__ __bf16 bf_cvt(float f) { return (__bf16)f; }
__device__ __forceinline__ unsigned short bf_bits(float f) {
    return __builtin_bit_cast(unsigned short, (__bf16)f);
}

// ---- Tensor Data Mover: 2-D tile load, 2-byte elements ---------------------
// D# group0: [1:0]=count=1, [63:32]=lds_addr, [120:64]=global_addr, [127:126]=2
// D# group1: [17:16]=data_size(1->2B), [79:48]=tensor_dim0, [111:80]=tensor_dim1,
//            [127:112]=tile_dim0, [143:128]=tile_dim1, [207:160]=dim0_stride
// Toolchain here is the 6-arg builtin: (g0, g1, g2, g3, g4, cpol).
__device__ __forceinline__ void tdm_load_2d(unsigned lds_addr, const void* gptr,
                                            unsigned tensor_d0, unsigned tensor_d1,
                                            unsigned tile_d0, unsigned tile_d1,
                                            unsigned stride0) {
    unsigned long long ga = (unsigned long long)(size_t)gptr;
    u32x4 g0;
    g0[0] = 1u;                                      // count=1, is_restore=0
    g0[1] = lds_addr;                                // LDS byte address
    g0[2] = (unsigned)ga;                            // global addr [31:0]
    g0[3] = (unsigned)(ga >> 32) | (2u << 30);       // addr [56:32] | type=2
    i32x8 g1;
    g1[0] = (int)(1u << 16);                         // mask=0, data_size=1 (2B)
    g1[1] = (int)((tensor_d0 & 0xFFFFu) << 16);      // tensor_dim0[15:0]
    g1[2] = (int)((tensor_d0 >> 16) | ((tensor_d1 & 0xFFFFu) << 16));
    g1[3] = (int)((tensor_d1 >> 16) | (tile_d0 << 16));
    g1[4] = (int)tile_d1;                            // tile_dim2 = 0
    g1[5] = (int)stride0;                            // dim0 stride [31:0]
    g1[6] = 0;
    g1[7] = 0;
    i32x4 z4 = {0, 0, 0, 0};                         // 2-D: groups 2/3 unused
    i32x8 z8 = {0, 0, 0, 0, 0, 0, 0, 0};
    __builtin_amdgcn_tensor_load_to_lds(g0, g1, z4, z4, z8, 0);
}

// --------------------------- phase 0: weight prep ---------------------------

// W [INS, OUTS] row-major fp32  ->  WT [OUTS, INS] bf16 (column-contiguous)
__global__ void cvt_transpose_w(const float* __restrict__ w,
                                unsigned short* __restrict__ wt) {
    int i = blockIdx.x * blockDim.x + threadIdx.x;
    if (i < INS * OUTS) {
        int k = i / OUTS, o = i % OUTS;
        wt[o * INS + k] = bf_bits(w[i]);
    }
}

// --------------------------- phase 1: QKV projection ------------------------
// One wave computes one 16-row tile of Q, K, V (each 16x64).  X is read as
// fp32 (single pass over 48MB) and converted in registers.  The W k-chunks
// (3 x 64x32 bf16 = 12KB) are TDM-staged into LDS, shared by all 8 waves.
// A (16x32 bf16) lane layout: row = lane%16, K = 16h+e
// B (32x16 bf16) lane layout: col = lane%16, K = 16h+e  -> 32B LDS read
// D (16x16 f32):  lane holds col = lane%16, rows r+8h across 8 VGPRs

__global__ void qkv_proj_wmma(const float* __restrict__ X,
                              const unsigned short* __restrict__ WqT,
                              const unsigned short* __restrict__ WkT,
                              const unsigned short* __restrict__ WvT,
                              const float* __restrict__ bq,
                              const float* __restrict__ bk,
                              const float* __restrict__ bv,
                              unsigned short* __restrict__ Qb,
                              unsigned short* __restrict__ Kb,
                              unsigned short* __restrict__ VbT) {
    // double-buffered W chunk: [buf][matrix][64 cols x 32 k] bf16  (24KB)
    __shared__ __attribute__((aligned(64))) unsigned short sW[2][3][64 * 32];

    const int lane = threadIdx.x & 31;
    const int wave = threadIdx.x >> 5;
    const int tile = blockIdx.x * (blockDim.x >> 5) + wave;   // 0..1023
    const int l16  = lane & 15;
    const int h    = lane >> 4;

    v8f acc[12];                                     // [matrix*4 + nb]
#pragma unroll
    for (int t = 0; t < 12; ++t)
#pragma unroll
        for (int r = 0; r < 8; ++r) acc[t][r] = 0.f;

    const float* xrow = X + (size_t)(tile * 16 + l16) * INS;

    // Prologue: TDM stages W chunk 0; pipeline the first A loads
    if (threadIdx.x < 32) {
        tdm_load_2d((unsigned)(size_t)&sW[0][0][0], WqT, INS, OUTS, 32, OUTS, INS);
        tdm_load_2d((unsigned)(size_t)&sW[0][1][0], WkT, INS, OUTS, 32, OUTS, INS);
        tdm_load_2d((unsigned)(size_t)&sW[0][2][0], WvT, INS, OUTS, 32, OUTS, INS);
        __builtin_amdgcn_s_wait_tensorcnt(0);
    }
    v8f f0 = *(const v8f*)(xrow + 16 * h);
    v8f f1 = *(const v8f*)(xrow + 8 + 16 * h);
    __syncthreads();

    const int nchunk = INS / 32;                     // 24
    // Rolled: the dynamic buffer index is one register add; unrolling here
    // overlaps live ranges at the seam and spills the 96-VGPR accumulators.
#pragma unroll 1
    for (int i = 0; i < nchunk; ++i) {
        const int cur = i & 1;

        // TDM prefetch of the next W chunk into the other buffer
        if ((i + 1) < nchunk && threadIdx.x < 32) {
            const int nxt = cur ^ 1;
            const int k2  = (i + 1) * 32;
            tdm_load_2d((unsigned)(size_t)&sW[nxt][0][0], WqT + k2, INS, OUTS, 32, OUTS, INS);
            tdm_load_2d((unsigned)(size_t)&sW[nxt][1][0], WkT + k2, INS, OUTS, 32, OUTS, INS);
            tdm_load_2d((unsigned)(size_t)&sW[nxt][2][0], WvT + k2, INS, OUTS, 32, OUTS, INS);
        }

        // Convert this chunk's A tile (hw packed cvt); issue next X loads
        v16bf a;
#pragma unroll
        for (int e = 0; e < 8; ++e) {
            a[e]     = bf_cvt(f0[e]);
            a[e + 8] = bf_cvt(f1[e]);
        }
        if ((i + 1) < nchunk) {
            const int k2 = (i + 1) * 32;
            f0 = *(const v8f*)(xrow + k2 + 16 * h);
            f1 = *(const v8f*)(xrow + k2 + 8 + 16 * h);
        }

        // 12 WMMAs; B-tiles are 32B ds reads, co-executing with the XDL pipe
        const unsigned short* wbase = &sW[cur][0][0];
#pragma unroll
        for (int mi = 0; mi < 3; ++mi)
#pragma unroll
            for (int nb = 0; nb < 4; ++nb) {
                v16bf bw = *(const v16bf*)(wbase + mi * (64 * 32) +
                                           (nb * 16 + l16) * 32 + 16 * h);
                acc[mi * 4 + nb] = __builtin_amdgcn_wmma_f32_16x16x32_bf16(
                    false, a, false, bw, (short)0, acc[mi * 4 + nb], false, false);
            }

        // next chunk has landed; everyone done reading buf[cur]
        if (threadIdx.x < 32) __builtin_amdgcn_s_wait_tensorcnt(0);
        __syncthreads();
    }

    const int grow0 = tile * 16;
#pragma unroll
    for (int nb = 0; nb < 4; ++nb) {
        const int col = nb * 16 + l16;
        const float vbq = bq[col], vbk = bk[col], vbv = bv[col];
#pragma unroll
        for (int r = 0; r < 8; ++r) {
            const int g = grow0 + r + 8 * h;                       // global row
            Qb[(size_t)g * OUTS + col] = bf_bits(acc[0 + nb][r] + vbq);
            Kb[(size_t)g * OUTS + col] = bf_bits(acc[4 + nb][r] + vbk);
            const int bb = g >> 11, ss = g & (SEQ - 1);            // V stored transposed
            VbT[(size_t)bb * OUTS * SEQ + (size_t)col * SEQ + ss] =
                bf_bits(acc[8 + nb][r] + vbv);
        }
    }
}

// --------------------------- phase 2: flash attention -----------------------
// 8 waves/block share one batch; wave w owns query tile (blockIdx%16)*8+w.
// K/V tiles for 32 keys are staged into LDS by the TDM (wave 0 issues the
// descriptors), double-buffered, one barrier per iteration.

__global__ void attention_wmma(const unsigned short* __restrict__ Qb,
                               const unsigned short* __restrict__ Kb,
                               const unsigned short* __restrict__ VbT,
                               float* __restrict__ out) {
    // double-buffered tiles: K tile 32x64 bf16, V^T tile 64x32 bf16 (4KB each)
    __shared__ __attribute__((aligned(64))) unsigned short sK[2][32 * 64];
    __shared__ __attribute__((aligned(64))) unsigned short sV[2][64 * 32];

    const int lane = threadIdx.x & 31;
    const int wave = threadIdx.x >> 5;
    const int b    = blockIdx.x >> 4;                         // batch
    const int qt   = (blockIdx.x & 15) * 8 + wave;            // query tile
    const int l16  = lane & 15;
    const int h    = lane >> 4;

    // Q tile as B-matrix chunks (dims 0..31 and 32..63), contiguous 32B loads
    const unsigned short* qrow = Qb + (size_t)(b * SEQ + qt * 16 + l16) * OUTS;
    const v16bf bq0 = *(const v16bf*)(qrow + 0 + 16 * h);
    const v16bf bq1 = *(const v16bf*)(qrow + 32 + 16 * h);

    const unsigned short* kbase = Kb  + (size_t)b * SEQ * OUTS;
    const unsigned short* vbase = VbT + (size_t)b * OUTS * SEQ;

    v8f o[4];
#pragma unroll
    for (int t = 0; t < 4; ++t)
#pragma unroll
        for (int r = 0; r < 8; ++r) o[t][r] = 0.f;

    float m = -__builtin_inff();
    float ssum = 0.f;
    const float scale = 0.125f;   // 1/sqrt(64)

    // Prologue: TDM stages tile 0 into buffer 0
    if (threadIdx.x < 32) {
        tdm_load_2d((unsigned)(size_t)&sK[0][0], kbase, OUTS, SEQ, OUTS, 32, OUTS);
        tdm_load_2d((unsigned)(size_t)&sV[0][0], vbase, SEQ, OUTS, 32, OUTS, SEQ);
        __builtin_amdgcn_s_wait_tensorcnt(0);
    }
    __syncthreads();

    const int niter = SEQ / 32;
    // Unroll by 2 so cur/nxt are compile-time constants -> LDS addresses
    // become immediates in the hot loop.
#pragma unroll 2
    for (int i = 0; i < niter; ++i) {
        const int j   = i * 32;
        const int cur = i & 1;

        // TDM prefetch of the next tile into the other buffer
        if ((i + 1) < niter && threadIdx.x < 32) {
            const int nxt = cur ^ 1;
            tdm_load_2d((unsigned)(size_t)&sK[nxt][0],
                        kbase + (size_t)(j + 32) * OUTS, OUTS, SEQ, OUTS, 32, OUTS);
            tdm_load_2d((unsigned)(size_t)&sV[nxt][0],
                        vbase + (j + 32), SEQ, OUTS, 32, OUTS, SEQ);
        }

        // ---- K tiles as A matrices from LDS (keys j..j+15, j+16..j+31) ----
        const unsigned short* k0 = &sK[cur][(size_t)l16 * OUTS];
        const unsigned short* k1 = &sK[cur][(size_t)(16 + l16) * OUTS];
        v16bf a00, a01, a10, a11;
        {
            v8bf t0 = *(const v8bf*)(k0 + 16 * h);
            v8bf t1 = *(const v8bf*)(k0 + 8 + 16 * h);
            v8bf t2 = *(const v8bf*)(k0 + 32 + 16 * h);
            v8bf t3 = *(const v8bf*)(k0 + 40 + 16 * h);
            v8bf u0 = *(const v8bf*)(k1 + 16 * h);
            v8bf u1 = *(const v8bf*)(k1 + 8 + 16 * h);
            v8bf u2 = *(const v8bf*)(k1 + 32 + 16 * h);
            v8bf u3 = *(const v8bf*)(k1 + 40 + 16 * h);
#pragma unroll
            for (int e = 0; e < 8; ++e) {
                a00[e] = t0[e]; a00[e + 8] = t1[e];
                a01[e] = t2[e]; a01[e + 8] = t3[e];
                a10[e] = u0[e]; a10[e + 8] = u1[e];
                a11[e] = u2[e]; a11[e + 8] = u3[e];
            }
        }

        // ---- score tiles: S^T[key][query], f32 accumulate ----
        v8f s0, s1;
#pragma unroll
        for (int r = 0; r < 8; ++r) { s0[r] = 0.f; s1[r] = 0.f; }
        s0 = __builtin_amdgcn_wmma_f32_16x16x32_bf16(false, a00, false, bq0, (short)0, s0, false, false);
        s0 = __builtin_amdgcn_wmma_f32_16x16x32_bf16(false, a01, false, bq1, (short)0, s0, false, false);
        s1 = __builtin_amdgcn_wmma_f32_16x16x32_bf16(false, a10, false, bq0, (short)0, s1, false, false);
        s1 = __builtin_amdgcn_wmma_f32_16x16x32_bf16(false, a11, false, bq1, (short)0, s1, false, false);

        // ---- online softmax (lane owns query l16; holds 16 of 32 keys) ----
        float tmax = -__builtin_inff();
#pragma unroll
        for (int r = 0; r < 8; ++r) {
            s0[r] *= scale; s1[r] *= scale;
            tmax = fmaxf(tmax, fmaxf(s0[r], s1[r]));
        }
        tmax = fmaxf(tmax, __shfl_xor(tmax, 16, 32));
        const float mnew = fmaxf(m, tmax);
        const float c = __expf(m - mnew);          // 0 on first iteration

        float p0[8], p1[8];
        float lsum = 0.f;
#pragma unroll
        for (int r = 0; r < 8; ++r) {
            p0[r] = __expf(s0[r] - mnew);
            p1[r] = __expf(s1[r] - mnew);
            lsum += p0[r] + p1[r];
        }
        lsum += __shfl_xor(lsum, 16, 32);
        ssum = ssum * c + lsum;
        m = mnew;

        // ---- P as A matrix (16 queries x 32 keys): layouts compose exactly
        v16bf ap;
#pragma unroll
        for (int e = 0; e < 8; ++e) {
            ap[e]     = bf_cvt(p0[e]);
            ap[e + 8] = bf_cvt(p1[e]);
        }

        // ---- rescale O rows by c (row = query r+8h -> bpermute broadcast)
        float cr[8];
#pragma unroll
        for (int r = 0; r < 8; ++r) cr[r] = __shfl(c, r + 8 * h, 32);
#pragma unroll
        for (int t = 0; t < 4; ++t)
#pragma unroll
            for (int r = 0; r < 8; ++r) o[t][r] *= cr[r];

        // ---- O += P x V  (V^T tile from LDS: two 16B B-tile loads, keys 8h+e / 16+8h+e)
#pragma unroll
        for (int t = 0; t < 4; ++t) {
            const int dim = t * 16 + l16;
            v8bf bv0 = *(const v8bf*)(&sV[cur][(size_t)dim * 32 + 8 * h]);
            v8bf bv1 = *(const v8bf*)(&sV[cur][(size_t)dim * 32 + 16 + 8 * h]);
            v16bf bv = __builtin_shufflevector(bv0, bv1, 0, 1, 2, 3, 4, 5, 6, 7, 8, 9, 10, 11, 12, 13, 14, 15);
            o[t] = __builtin_amdgcn_wmma_f32_16x16x32_bf16(
                false, ap, false, bv, (short)0, o[t], false, false);
        }

        // next tile has landed; everyone done reading buf[cur]
        if (threadIdx.x < 32) __builtin_amdgcn_s_wait_tensorcnt(0);
        __syncthreads();
    }

    // ---- normalize and store (row = query r+8h, col = t*16+l16) ----
    const float inv = 1.0f / ssum;
    float ir[8];
#pragma unroll
    for (int r = 0; r < 8; ++r) ir[r] = __shfl(inv, r + 8 * h, 32);

    const int grow0 = b * SEQ + qt * 16;
#pragma unroll
    for (int t = 0; t < 4; ++t) {
        const int col = t * 16 + l16;
#pragma unroll
        for (int r = 0; r < 8; ++r) {
            out[(size_t)(grow0 + r + 8 * h) * OUTS + col] = o[t][r] * ir[r];
        }
    }
}

// --------------------------------- launcher ---------------------------------

extern "C" void kernel_launch(void* const* d_in, const int* in_sizes, int n_in,
                              void* d_out, int out_size, void* d_ws, size_t ws_size,
                              hipStream_t stream) {
    (void)in_sizes; (void)n_in; (void)out_size; (void)ws_size;

    const float* X  = (const float*)d_in[0];
    const float* Wq = (const float*)d_in[1];
    const float* bq = (const float*)d_in[2];
    const float* Wk = (const float*)d_in[3];
    const float* bk = (const float*)d_in[4];
    const float* Wv = (const float*)d_in[5];
    const float* bv = (const float*)d_in[6];
    float* out = (float*)d_out;

    char* ws = (char*)d_ws;
    size_t off = 0;
    auto alloc = [&](size_t bytes) -> void* {
        void* p = ws + off;
        off = (off + bytes + 255) & ~(size_t)255;
        return p;
    };

    unsigned short* WqT = (unsigned short*)alloc((size_t)INS * OUTS * 2);
    unsigned short* WkT = (unsigned short*)alloc((size_t)INS * OUTS * 2);
    unsigned short* WvT = (unsigned short*)alloc((size_t)INS * OUTS * 2);
    unsigned short* Qb  = (unsigned short*)alloc((size_t)BATCH * SEQ * OUTS * 2);
    unsigned short* Kb  = (unsigned short*)alloc((size_t)BATCH * SEQ * OUTS * 2);
    unsigned short* VbT = (unsigned short*)alloc((size_t)BATCH * SEQ * OUTS * 2);

    // Phase 0: weight prep (tiny; L2-resident afterwards)
    const int nW = INS * OUTS;
    cvt_transpose_w<<<(nW + 255) / 256, 256, 0, stream>>>(Wq, WqT);
    cvt_transpose_w<<<(nW + 255) / 256, 256, 0, stream>>>(Wk, WkT);
    cvt_transpose_w<<<(nW + 255) / 256, 256, 0, stream>>>(Wv, WvT);

    // Phase 1: QKV projection — 1024 row-tiles, 8 waves/block; X read once
    qkv_proj_wmma<<<128, 256, 0, stream>>>(X, WqT, WkT, WvT, bq, bk, bv,
                                           Qb, Kb, VbT);

    // Phase 2: attention — 8 query tiles per block sharing one batch's K/V
    attention_wmma<<<128, 256, 0, stream>>>(Qb, Kb, VbT, out);
}
